// SelfAttention_6347961664048
// MI455X (gfx1250) — matrix operand order repaired and checked by reading, not compiled, not run on hardware
//
#include <hip/hip_runtime.h>

// ---------------------------------------------------------------------------
// Self-attention for MI455X (gfx1250, wave32, WMMA + async-to-LDS).
// B=8, S=1024, D=1024, H=16, HD=64.
// Pipeline: 3x projection GEMM (f32 in -> f16 LDS tiles -> wmma f32 acc,
//           register-prefetch double buffering) then flash-attention with
//           GLOBAL_LOAD_ASYNC_TO_LDS_B128 double-buffered K/V tiles.
// ---------------------------------------------------------------------------

#define BATCH 8
#define SEQ   1024
#define DIM   1024
#define NH    16
#define HD    64
#define NEG_BIG 1e12f

typedef _Float16 h16 __attribute__((ext_vector_type(16)));
typedef _Float16 h4  __attribute__((ext_vector_type(4)));
typedef float    f8  __attribute__((ext_vector_type(8)));

union H16u { h16 v; h4 q[4]; };

// ---- WMMA wrapper: D = A(16x32 f16) * B(32x16 f16) + C(16x16 f32) ----------
__device__ __forceinline__ f8 wmma16(h16 a, h16 b, f8 c) {
  return __builtin_amdgcn_wmma_f32_16x16x32_f16(
      /*neg_a=*/false, a, /*neg_b=*/false, b,
      /*c_mod=*/(short)0, c, /*reuse_a=*/false, /*reuse_b=*/false);
}

// ---- A-fragment (16x32, MxK) from LDS, row-major [row][k], stride ld -------
// lane l (r=l&15, hi=l>>4): elems 0..7 -> K = 8*hi + 0..7
//                           elems 8..15 -> K = 16 + 8*hi + 0..7
__device__ __forceinline__ h16 ldsA(const _Float16* base, int ld) {
  const int lane = threadIdx.x & 31;
  const int r = lane & 15, hi = lane >> 4;
  const _Float16* p = base + r * ld + 8 * hi;
  H16u a;
  a.q[0] = *(const h4*)(p + 0);
  a.q[1] = *(const h4*)(p + 4);
  a.q[2] = *(const h4*)(p + 16);
  a.q[3] = *(const h4*)(p + 20);
  return a.v;
}

// ---- B-fragment (32x16, KxN) from LDS stored transposed [n][k], stride ld --
// lane l (c=l&15, hi=l>>4): elems 0..7 -> K = 8*hi + 0..7,
//                           elems 8..15 -> K = 16 + 8*hi + 0..7, column N = c
__device__ __forceinline__ h16 ldsB(const _Float16* base, int ld) {
  const int lane = threadIdx.x & 31;
  const int c = lane & 15, hi = lane >> 4;
  const _Float16* p = base + c * ld + 8 * hi;
  H16u b;
  b.q[0] = *(const h4*)(p + 0);
  b.q[1] = *(const h4*)(p + 4);
  b.q[2] = *(const h4*)(p + 16);
  b.q[3] = *(const h4*)(p + 20);
  return b.v;
}

// ---- CDNA5 async global->LDS copy (ASYNCcnt-tracked, no VGPR staging) ------
// lds_byte_addr: LDS offset == low 32 bits of the flat address of a __shared__
// object (ISA 10.2).
__device__ __forceinline__ void async_ld_b128(unsigned lds_byte_addr,
                                              const void* gaddr) {
  asm volatile("global_load_async_to_lds_b128 %0, %1, off"
               :: "v"(lds_byte_addr), "v"(gaddr)
               : "memory");
}
__device__ __forceinline__ void wait_async_le8() {
  asm volatile("s_wait_asynccnt 0x8" ::: "memory");
}
__device__ __forceinline__ void wait_async_0() {
  asm volatile("s_wait_asynccnt 0x0" ::: "memory");
}
__device__ __forceinline__ unsigned lds_addr_of(const void* p) {
  return (unsigned)(size_t)p;
}

// ===========================================================================
// Projection GEMM: out[m,n] = sum_k X[m,k] * W[k,n];  M=8192, N=K=1024.
// Output f16, scattered to [B,H,S,HD] (vtranspose=0) or [B,H,HD,S] (=1).
// Block: 256 thr (8 waves), tile 128x128x32, wave tile 32x64.
// Register-prefetch double buffering: next k-slice global loads overlap WMMA.
// ===========================================================================
#define PBM 128
#define PBK 32
#define PLDA 40   // 32 + 8 pad (halfs); keeps 8B alignment for h4 loads
#define PLDB 40

__global__ __launch_bounds__(256) void proj_kernel(
    const float* __restrict__ X, const float* __restrict__ W,
    _Float16* __restrict__ out, int vtranspose)
{
  __shared__ _Float16 lA[PBM * PLDA];   // A tile, row-major [m][k]
  __shared__ _Float16 lB[PBM * PLDB];   // B tile, transposed [n][k]

  const int tid  = threadIdx.x;
  const int lane = tid & 31;
  const int wid  = tid >> 5;
  const int wm   = wid & 3;   // 4 waves along M (32 rows each)
  const int wn   = wid >> 2;  // 2 waves along N (64 cols each)
  const int m0   = blockIdx.x * PBM;
  const int n0   = blockIdx.y * PBM;

  f8 acc[2][4];
#pragma unroll
  for (int i = 0; i < 2; ++i)
#pragma unroll
    for (int j = 0; j < 4; ++j)
#pragma unroll
      for (int v = 0; v < 8; ++v) acc[i][j][v] = 0.0f;

  float4 ra[4], rb[4];

  // ---- prologue: fetch k-slice 0 into registers, commit to LDS ----
#pragma unroll
  for (int i = 0; i < 4; ++i) {
    const int idx = tid + 256 * i;
    ra[i] = *(const float4*)(X + (size_t)(m0 + (idx >> 3)) * DIM + 4 * (idx & 7));
    rb[i] = *(const float4*)(W + (size_t)(idx >> 5) * DIM + n0 + 4 * (idx & 31));
  }
#pragma unroll
  for (int i = 0; i < 4; ++i) {
    const int idx = tid + 256 * i;
    {
      const int r = idx >> 3, c4 = idx & 7;
      _Float16* d = &lA[r * PLDA + 4 * c4];
      d[0] = (_Float16)ra[i].x; d[1] = (_Float16)ra[i].y;
      d[2] = (_Float16)ra[i].z; d[3] = (_Float16)ra[i].w;
    }
    {
      const int kk = idx >> 5, c4 = idx & 31;
      lB[(4 * c4 + 0) * PLDB + kk] = (_Float16)rb[i].x;
      lB[(4 * c4 + 1) * PLDB + kk] = (_Float16)rb[i].y;
      lB[(4 * c4 + 2) * PLDB + kk] = (_Float16)rb[i].z;
      lB[(4 * c4 + 3) * PLDB + kk] = (_Float16)rb[i].w;
    }
  }

  int k0 = PBK;
  for (;;) {
    __syncthreads();                    // LDS tile visible to all waves
    const bool more = (k0 < DIM);
    if (more) {                         // prefetch next slice during compute
#pragma unroll
      for (int i = 0; i < 4; ++i) {
        const int idx = tid + 256 * i;
        ra[i] = *(const float4*)(X + (size_t)(m0 + (idx >> 3)) * DIM + k0 + 4 * (idx & 7));
        rb[i] = *(const float4*)(W + (size_t)(k0 + (idx >> 5)) * DIM + n0 + 4 * (idx & 31));
      }
    }

    h16 af[2];
#pragma unroll
    for (int mi = 0; mi < 2; ++mi)
      af[mi] = ldsA(&lA[(wm * 32 + mi * 16) * PLDA], PLDA);
#pragma unroll
    for (int ni = 0; ni < 4; ++ni) {
      const h16 bf = ldsB(&lB[(wn * 64 + ni * 16) * PLDB], PLDB);
#pragma unroll
      for (int mi = 0; mi < 2; ++mi)
        acc[mi][ni] = wmma16(af[mi], bf, acc[mi][ni]);
    }

    if (!more) break;
    __syncthreads();                    // all waves done reading LDS
#pragma unroll
    for (int i = 0; i < 4; ++i) {
      const int idx = tid + 256 * i;
      {
        const int r = idx >> 3, c4 = idx & 7;
        _Float16* d = &lA[r * PLDA + 4 * c4];
        d[0] = (_Float16)ra[i].x; d[1] = (_Float16)ra[i].y;
        d[2] = (_Float16)ra[i].z; d[3] = (_Float16)ra[i].w;
      }
      {
        const int kk = idx >> 5, c4 = idx & 31;
        lB[(4 * c4 + 0) * PLDB + kk] = (_Float16)rb[i].x;
        lB[(4 * c4 + 1) * PLDB + kk] = (_Float16)rb[i].y;
        lB[(4 * c4 + 2) * PLDB + kk] = (_Float16)rb[i].z;
        lB[(4 * c4 + 3) * PLDB + kk] = (_Float16)rb[i].w;
      }
    }
    k0 += PBK;
  }

  // --- epilogue: scatter f32 acc -> f16 output in head-major layout ---
  const int hi = lane >> 4, c = lane & 15;
#pragma unroll
  for (int mi = 0; mi < 2; ++mi) {
#pragma unroll
    for (int ni = 0; ni < 4; ++ni) {
#pragma unroll
      for (int v = 0; v < 8; ++v) {
        const int m = m0 + wm * 32 + mi * 16 + v + 8 * hi;
        const int n = n0 + wn * 64 + ni * 16 + c;
        const int b = m >> 10, s = m & (SEQ - 1);
        const int h = n >> 6,  hd = n & (HD - 1);
        const size_t o = vtranspose
            ? ((size_t)(b * NH + h) * HD + hd) * SEQ + s
            : ((size_t)(b * NH + h) * SEQ + s) * HD + hd;
        out[o] = (_Float16)acc[mi][ni][v];
      }
    }
  }
}

// ===========================================================================
// Flash attention: one block per (b*H+h, 128-query tile). 8 waves; each wave
// owns 16 queries. K/V tiles double-buffered via async global->LDS copies:
// prefetch of tile kt+1 overlaps all WMMA/softmax work of tile kt.
// ===========================================================================
#define AQ   128
#define AKV  128
#define LQK  72    // 64 + 8 pad (halfs)
#define LV   136   // 128 + 8 pad
#define LP   136
#define KBUF (AKV * LQK)
#define VBUF (HD * LV)

__device__ __forceinline__ float half_red_max(float x) {
#pragma unroll
  for (int m = 1; m <= 8; m <<= 1) x = fmaxf(x, __shfl_xor(x, m, 32));
  return x;  // reduced within each 16-lane half
}
__device__ __forceinline__ float half_red_sum(float x) {
#pragma unroll
  for (int m = 1; m <= 8; m <<= 1) x += __shfl_xor(x, m, 32);
  return x;
}

// Issue one K/V tile's async loads: 8 x b128 per thread (4 K + 4 V).
__device__ __forceinline__ void issue_tile(
    const _Float16* kb, const _Float16* vb, int k0,
    unsigned sK_base, unsigned sV_base, int tid)
{
#pragma unroll
  for (int i = 0; i < 4; ++i) {
    const int idx = tid + 256 * i;          // 1024 x 16B chunks
    const int r = idx >> 3, c8 = idx & 7;   // [key][hd]
    async_ld_b128(sK_base + 2u * (unsigned)(r * LQK + c8 * 8),
                  kb + (size_t)(k0 + r) * HD + c8 * 8);
  }
#pragma unroll
  for (int i = 0; i < 4; ++i) {
    const int idx = tid + 256 * i;
    const int r = idx >> 4, c8 = idx & 15;  // [hd][key]
    async_ld_b128(sV_base + 2u * (unsigned)(r * LV + c8 * 8),
                  vb + (size_t)r * SEQ + k0 + c8 * 8);
  }
}

__global__ __launch_bounds__(256) void attn_kernel(
    const _Float16* __restrict__ qh, const _Float16* __restrict__ kh,
    const _Float16* __restrict__ vh, const float* __restrict__ masks,
    float* __restrict__ out)
{
  __shared__ _Float16 sQ[AQ * LQK];        // [query][hd]
  __shared__ _Float16 sK[2 * KBUF];        // [key][hd]  double-buffered
  __shared__ _Float16 sV[2 * VBUF];        // [hd][key]  double-buffered
  __shared__ _Float16 sP[8 * 16 * LP];     // per-wave probability staging

  const int tid  = threadIdx.x;
  const int lane = tid & 31;
  const int wid  = tid >> 5;
  const int hi   = lane >> 4, c = lane & 15;
  const int q0   = blockIdx.x * AQ;
  const int bh   = blockIdx.y;                 // b*NH + h
  const int b    = bh >> 4, h = bh & (NH - 1);

  const _Float16* qb = qh + (size_t)bh * SEQ * HD;
  const _Float16* kb = kh + (size_t)bh * SEQ * HD;
  const _Float16* vb = vh + (size_t)bh * HD * SEQ;
  const float*    mk = masks + (size_t)b * SEQ;

  const unsigned sQb = lds_addr_of(&sQ[0]);
  const unsigned sKb0 = lds_addr_of(&sK[0]);
  const unsigned sVb0 = lds_addr_of(&sV[0]);

  // --- async-load Q tile (4 x b128/thread), then K/V tile 0 into buffer 0 ---
#pragma unroll
  for (int i = 0; i < 4; ++i) {
    const int idx = tid + 256 * i;
    const int r = idx >> 3, c8 = idx & 7;
    async_ld_b128(sQb + 2u * (unsigned)(r * LQK + c8 * 8),
                  qb + (size_t)(q0 + r) * HD + c8 * 8);
  }
  issue_tile(kb, vb, 0, sKb0, sVb0, tid);   // 12 outstanding (4 Q + 8 tile0)
  wait_async_le8();                         // in-order completion -> Q landed
  __syncthreads();

  h16 qf[2];
  qf[0] = ldsA(&sQ[(wid * 16) * LQK +  0], LQK);
  qf[1] = ldsA(&sQ[(wid * 16) * LQK + 32], LQK);

  f8 O[4];
#pragma unroll
  for (int i = 0; i < 4; ++i)
#pragma unroll
    for (int v = 0; v < 8; ++v) O[i][v] = 0.0f;
  float mrow[8], lrow[8];
#pragma unroll
  for (int v = 0; v < 8; ++v) { mrow[v] = -__builtin_inff(); lrow[v] = 0.0f; }

  _Float16* pw = &sP[wid * 16 * LP];

  for (int kt = 0; kt < SEQ / AKV; ++kt) {
    const int cur = kt & 1;
    const _Float16* sKc = &sK[cur * KBUF];
    const _Float16* sVc = &sV[cur * VBUF];
    const int k0 = kt * AKV;
    const bool pre = (kt + 1) < (SEQ / AKV);

    // prefetch next tile into the other buffer (freed by last trailing barrier)
    if (pre) issue_tile(kb, vb, k0 + AKV,
                        sKb0 + (unsigned)((cur ^ 1) * KBUF * 2),
                        sVb0 + (unsigned)((cur ^ 1) * VBUF * 2), tid);
    // async loads complete in order: <=8 outstanding => current tile landed
    if (pre) wait_async_le8(); else wait_async_0();
    __syncthreads();

    // --- scores: S = Q * K^T / sqrt(HD) + key-padding mask ---
    f8 Sc[8];
#pragma unroll
    for (int nt = 0; nt < 8; ++nt) {
      f8 s;
#pragma unroll
      for (int v = 0; v < 8; ++v) s[v] = 0.0f;
#pragma unroll
      for (int ks = 0; ks < 2; ++ks) {
        const h16 bf = ldsB(&sKc[(nt * 16) * LQK + ks * 32], LQK);
        s = wmma16(qf[ks], bf, s);
      }
      const float addv = (1.0f - mk[k0 + nt * 16 + c]) * (-NEG_BIG);
#pragma unroll
      for (int v = 0; v < 8; ++v) s[v] = s[v] * 0.125f + addv;
      Sc[nt] = s;
    }

    // --- online softmax (row = v + 8*hi lives in one 16-lane half) ---
    float alpha[8];
#pragma unroll
    for (int v = 0; v < 8; ++v) {
      float mx = Sc[0][v];
#pragma unroll
      for (int nt = 1; nt < 8; ++nt) mx = fmaxf(mx, Sc[nt][v]);
      mx = half_red_max(mx);
      const float mn = fmaxf(mrow[v], mx);
      alpha[v] = __expf(mrow[v] - mn);
      mrow[v] = mn;
      float rs = 0.0f;
#pragma unroll
      for (int nt = 0; nt < 8; ++nt) {
        const float p = __expf(Sc[nt][v] - mn);
        Sc[nt][v] = p;
        rs += p;
      }
      rs = half_red_sum(rs);
      lrow[v] = lrow[v] * alpha[v] + rs;
    }
#pragma unroll
    for (int nt = 0; nt < 4; ++nt)
#pragma unroll
      for (int v = 0; v < 8; ++v) O[nt][v] *= alpha[v];

    // --- stage P (f16) in A-layout [row][key] for the PV matmul ---
#pragma unroll
    for (int nt = 0; nt < 8; ++nt)
#pragma unroll
      for (int v = 0; v < 8; ++v)
        pw[(v + 8 * hi) * LP + nt * 16 + c] = (_Float16)Sc[nt][v];
    __syncthreads();

    // --- O += P * V ---
#pragma unroll
    for (int kk = 0; kk < 4; ++kk) {
      const h16 pf = ldsA(&pw[kk * 32], LP);
#pragma unroll
      for (int nt = 0; nt < 4; ++nt) {
        const h16 vf = ldsB(&sVc[(nt * 16) * LV + kk * 32], LV);
        O[nt] = wmma16(pf, vf, O[nt]);
      }
    }
    __syncthreads();   // all waves done with buf[cur] before it is refilled
  }

  // --- epilogue: normalize, apply query mask, write f32 output ---
  float wq[8];
#pragma unroll
  for (int v = 0; v < 8; ++v) {
    const int q = q0 + wid * 16 + v + 8 * hi;
    wq[v] = mk[q] / lrow[v];
  }
#pragma unroll
  for (int nt = 0; nt < 4; ++nt) {
#pragma unroll
    for (int v = 0; v < 8; ++v) {
      const int q = q0 + wid * 16 + v + 8 * hi;
      out[((size_t)(b * SEQ + q)) * (NH * HD) + h * HD + nt * 16 + c] =
          O[nt][v] * wq[v];
    }
  }
}

// ===========================================================================
extern "C" void kernel_launch(void* const* d_in, const int* in_sizes, int n_in,
                              void* d_out, int out_size, void* d_ws, size_t ws_size,
                              hipStream_t stream) {
  (void)in_sizes; (void)n_in; (void)out_size; (void)ws_size;
  const float* Qs = (const float*)d_in[0];
  const float* Ks = (const float*)d_in[1];
  const float* Vs = (const float*)d_in[2];
  const float* mk = (const float*)d_in[3];
  const float* WQ = (const float*)d_in[4];
  const float* WK = (const float*)d_in[5];
  const float* WV = (const float*)d_in[6];

  const size_t per = (size_t)BATCH * NH * SEQ * HD;   // 8.4M halfs = 16 MB each
  _Float16* qh = (_Float16*)d_ws;
  _Float16* kh = qh + per;
  _Float16* vh = kh + per;

  dim3 blk(256);
  dim3 gp(BATCH * SEQ / PBM, DIM / PBM, 1);           // 64 x 8
  proj_kernel<<<gp, blk, 0, stream>>>(Qs, WQ, qh, 0);
  proj_kernel<<<gp, blk, 0, stream>>>(Ks, WK, kh, 0);
  proj_kernel<<<gp, blk, 0, stream>>>(Vs, WV, vh, 1); // V stored [B,H,HD,S]

  dim3 ga(SEQ / AQ, BATCH * NH, 1);                   // 8 x 128
  attn_kernel<<<ga, blk, 0, stream>>>(qh, kh, vh, mk, (float*)d_out);
}
